// LSTMDecoder_15650860826806
// MI455X (gfx1250) — matrix operand order repaired and checked by reading, not compiled, not run on hardware
//
#include <hip/hip_runtime.h>
#include <hip/hip_bf16.h>
#include <math.h>

typedef __attribute__((ext_vector_type(16))) _Float16 v16h;
typedef __attribute__((ext_vector_type(8)))  _Float16 v8h;
typedef __attribute__((ext_vector_type(8)))  float    v8f;
typedef __attribute__((ext_vector_type(4)))  int      v4i;

#define B_DIM 2048
#define H_DIM 512
#define V_DIM 64
#define T_DIM 128
#define G4H   (4 * H_DIM)

#if __has_builtin(__builtin_amdgcn_global_load_async_to_lds_b128) && \
    __has_builtin(__builtin_amdgcn_s_wait_asynccnt)
#define USE_ASYNC_LDS 1
typedef __attribute__((address_space(1))) v4i* gptr_v4i;
typedef __attribute__((address_space(3))) v4i* lptr_v4i;
#endif

// ---------------------------------------------------------------------------
// WMMA fragment loaders, using contiguous k order (k = kg*16 + i) in both A and B, not the ISA table order.
//
// A (16x32 f16, 8 VGPRs): lane L: row m = L&15, kg = L>>4;
//   elements [0..7]  = A[m][kg*16 + 0..7]
//   elements [8..15] = A[m][kg*16 + 8 + 0..7]
// B (32x16 f16, 8 VGPRs): lane L: col n = L&15, kg = L>>4;
//   elements [i] = B[kg*16 + i][n] = W[n][kg*16 + i]   (B = W^T, contiguous)
// C/D (16x16 f32, 8 VGPRs): lane L: n = L&15; acc[v] = D[v + 8*(L>>4)][n]
// ---------------------------------------------------------------------------
__device__ __forceinline__ v16h load_a_frag(const _Float16* __restrict__ base,
                                            int lda, int m0, int k0, int lane) {
    int m  = m0 + (lane & 15);
    int kg = lane >> 4;
    const _Float16* p = base + (size_t)m * lda + k0 + kg * 16;
    v8h lo = *(const v8h*)(p);
    v8h hi = *(const v8h*)(p + 8);
    return __builtin_shufflevector(lo, hi, 0, 1, 2, 3, 4, 5, 6, 7,
                                           8, 9, 10, 11, 12, 13, 14, 15);
}

__device__ __forceinline__ v16h load_b_frag(const _Float16* __restrict__ base,
                                            int ldb, int n0, int k0, int lane) {
    int n  = n0 + (lane & 15);
    int kg = lane >> 4;
    const _Float16* p = base + (size_t)n * ldb + k0 + kg * 16;
    return *(const v16h*)(p);
}

__device__ __forceinline__ v8f wmma16x16x32(v16h a, v16h b, v8f c) {
    return __builtin_amdgcn_wmma_f32_16x16x32_f16(
        /*neg_a=*/false, a, /*neg_b=*/false, b,
        /*c_mod=*/(short)0, c, /*reuse_a=*/false, /*reuse_b=*/false);
}

__device__ __forceinline__ float fast_sigmoid(float x) {
    return 1.0f / (1.0f + __expf(-x));
}

// ---------------------------------------------------------------------------
// Utility kernels
// ---------------------------------------------------------------------------
__global__ void cvt_f32_f16(const float* __restrict__ s, _Float16* __restrict__ d, int n) {
    int i = blockIdx.x * blockDim.x + threadIdx.x;
    if (i < n) d[i] = (_Float16)s[i];
}

__global__ void init_state(float* __restrict__ c, _Float16* __restrict__ h, int n) {
    int i = blockIdx.x * blockDim.x + threadIdx.x;
    if (i < n) { c[i] = 0.0f; h[i] = (_Float16)0.0f; }
}

// ---------------------------------------------------------------------------
// x_gates = z @ w_ih^T + b_ih + b_hh      [B, 4H] f32  (runs once)
// ---------------------------------------------------------------------------
__global__ void xgates_gemm(const _Float16* __restrict__ zf,
                            const _Float16* __restrict__ wih,
                            const float* __restrict__ b_ih,
                            const float* __restrict__ b_hh,
                            float* __restrict__ xg) {
    int lane  = threadIdx.x & 31;
    int wv    = threadIdx.x >> 5;
    int gtile = blockIdx.x * 8 + wv;
    int m0 = (gtile >> 7) << 4;
    int n0 = (gtile & 127) << 4;

    v8f acc = {};
#pragma unroll
    for (int k0 = 0; k0 < H_DIM; k0 += 32) {
        v16h a = load_a_frag(zf,  H_DIM, m0, k0, lane);
        v16h b = load_b_frag(wih, H_DIM, n0, k0, lane);
        acc = wmma16x16x32(a, b, acc);
    }

    int n    = n0 + (lane & 15);
    int half = lane >> 4;
    float bias = b_ih[n] + b_hh[n];
#pragma unroll
    for (int v = 0; v < 8; ++v) {
        int m = m0 + v + 8 * half;
        xg[(size_t)m * G4H + n] = acc[v] + bias;
    }
}

// ---------------------------------------------------------------------------
// One LSTM step, fused: gates = x_gates + h_in @ w_hh^T, then cell update.
//
// Block = 8 waves, all sharing ONE j-tile: the 64 KB of w_hh rows they all
// need (4 gates x 16 rows x 512 K, f16) is staged into LDS once per block
// via gfx1250 async global->LDS copies, cutting w_hh L2 traffic 8x.
// Each wave then computes one 16(b) x 16(j) tile for ALL FOUR gates
// (A fragment of h shared across the 4 gate WMMAs; B fragments from LDS).
// Grid: 32 j-tiles x 16 m-groups = 512 blocks.
// ---------------------------------------------------------------------------
__global__ void __launch_bounds__(256)
lstm_step(const _Float16* __restrict__ h_in,
          const _Float16* __restrict__ whh,
          const float* __restrict__ xg,
          float* __restrict__ c,
          _Float16* __restrict__ h_out) {
    __shared__ __align__(32) _Float16 lw[4 * 16 * H_DIM];   // 64 KB

    int lane = threadIdx.x & 31;
    int wv   = threadIdx.x >> 5;
    int jt   = blockIdx.x & 31;    // 32 j-tiles
    int mg   = blockIdx.x >> 5;    // 16 m-groups of 8 m-tiles
    int j0   = jt << 4;
    int m0   = ((mg << 3) + wv) << 4;

    // ---- Stage w_hh rows {g*H + j0 .. +16 | g=0..3} into LDS ----
    // 4096 chunks of 16 B; 16 chunks per thread.
    {
        int tid = threadIdx.x;
#pragma unroll
        for (int it = 0; it < 16; ++it) {
            int ch   = tid + it * 256;
            int row  = ch >> 6;            // 0..63: gate*16 + r
            int off8 = (ch & 63) * 8;      // f16 element offset within row
            int gate = row >> 4;
            int r    = row & 15;
            const _Float16* src =
                whh + (size_t)(gate * H_DIM + j0 + r) * H_DIM + off8;
            _Float16* dst = &lw[(size_t)row * H_DIM + off8];
#ifdef USE_ASYNC_LDS
            __builtin_amdgcn_global_load_async_to_lds_b128(
                (gptr_v4i)(const_cast<_Float16*>(src)),
                (lptr_v4i)(dst),
                /*offset=*/0, /*cpol=*/0);
#else
            *(v8h*)dst = *(const v8h*)src;
#endif
        }
    }
#ifdef USE_ASYNC_LDS
    __builtin_amdgcn_s_wait_asynccnt(0);
#endif
    __syncthreads();

    // ---- GEMM: 4 gate accumulators, A from global (L2), B from LDS ----
    int n  = lane & 15;
    int kg = lane >> 4;
    v8f acc_i = {}, acc_f = {}, acc_g = {}, acc_o = {};
#pragma unroll
    for (int k0 = 0; k0 < H_DIM; k0 += 32) {
        v16h a = load_a_frag(h_in, H_DIM, m0, k0, lane);
        int kb = k0 + kg * 16;
        v16h bi = *(const v16h*)&lw[(0 * 16 + n) * H_DIM + kb];
        v16h bf = *(const v16h*)&lw[(1 * 16 + n) * H_DIM + kb];
        v16h bg = *(const v16h*)&lw[(2 * 16 + n) * H_DIM + kb];
        v16h bo = *(const v16h*)&lw[(3 * 16 + n) * H_DIM + kb];
        acc_i = wmma16x16x32(a, bi, acc_i);
        acc_f = wmma16x16x32(a, bf, acc_f);
        acc_g = wmma16x16x32(a, bg, acc_g);
        acc_o = wmma16x16x32(a, bo, acc_o);
    }

    // ---- Cell update: same (lane,vgpr) slot = same (b, j) in all accs ----
    int col  = j0 + n;
    int half = kg;
#pragma unroll
    for (int v = 0; v < 8; ++v) {
        int m = m0 + v + 8 * half;
        size_t gr = (size_t)m * G4H;
        float gi = acc_i[v] + xg[gr + 0 * H_DIM + col];
        float gf = acc_f[v] + xg[gr + 1 * H_DIM + col];
        float gg = acc_g[v] + xg[gr + 2 * H_DIM + col];
        float go = acc_o[v] + xg[gr + 3 * H_DIM + col];

        float ii = fast_sigmoid(gi);
        float ff = fast_sigmoid(gf);
        float g2 = tanhf(gg);
        float oo = fast_sigmoid(go);

        size_t ci = (size_t)m * H_DIM + col;
        float cc = ff * c[ci] + ii * g2;
        c[ci] = cc;
        h_out[ci] = (_Float16)(oo * tanhf(cc));
    }
}

// ---------------------------------------------------------------------------
// Per-step FC: out[b, t, :] = relu(h_t @ fc_w^T + fc_b)
// ---------------------------------------------------------------------------
__global__ void fc_step(const _Float16* __restrict__ h,
                        const _Float16* __restrict__ fcw,
                        const float* __restrict__ fcb,
                        float* __restrict__ out, int t) {
    int lane  = threadIdx.x & 31;
    int wv    = threadIdx.x >> 5;
    int gtile = blockIdx.x * 8 + wv;
    int m0 = (gtile >> 2) << 4;
    int n0 = (gtile & 3) << 4;

    v8f acc = {};
#pragma unroll
    for (int k0 = 0; k0 < H_DIM; k0 += 32) {
        v16h a = load_a_frag(h,   H_DIM, m0, k0, lane);
        v16h b = load_b_frag(fcw, H_DIM, n0, k0, lane);
        acc = wmma16x16x32(a, b, acc);
    }

    int n    = n0 + (lane & 15);
    int half = lane >> 4;
    float bias = fcb[n];
#pragma unroll
    for (int v = 0; v < 8; ++v) {
        int m = m0 + v + 8 * half;
        float r = acc[v] + bias;
        out[(size_t)m * (T_DIM * V_DIM) + (size_t)t * V_DIM + n] = r > 0.0f ? r : 0.0f;
    }
}

// ---------------------------------------------------------------------------
// Host launcher
// ---------------------------------------------------------------------------
extern "C" void kernel_launch(void* const* d_in, const int* in_sizes, int n_in,
                              void* d_out, int out_size, void* d_ws, size_t ws_size,
                              hipStream_t stream) {
    (void)in_sizes; (void)n_in; (void)out_size; (void)ws_size;

    const float* z    = (const float*)d_in[0];  // [B, H]
    const float* w_ih = (const float*)d_in[1];  // [4H, H]
    const float* w_hh = (const float*)d_in[2];  // [4H, H]
    const float* b_ih = (const float*)d_in[3];  // [4H]
    const float* b_hh = (const float*)d_in[4];  // [4H]
    const float* fc_w = (const float*)d_in[5];  // [V, H]
    const float* fc_b = (const float*)d_in[6];  // [V]
    float* out = (float*)d_out;                 // [B, T, V]

    char* ws = (char*)d_ws;
    const size_t MB = 1024 * 1024;
    _Float16* zf   = (_Float16*)(ws + 0 * MB);            // 2 MB
    _Float16* wihf = (_Float16*)(ws + 2 * MB);            // 2 MB
    _Float16* whhf = (_Float16*)(ws + 4 * MB);            // 2 MB
    _Float16* fcwf = (_Float16*)(ws + 6 * MB);            // 64 KB
    float*    xg   = (float*)   (ws + 6 * MB + 65536);    // 16 MB
    float*    c    = (float*)   (ws + 22 * MB + 65536);   // 4 MB
    _Float16* h0   = (_Float16*)(ws + 26 * MB + 65536);   // 2 MB
    _Float16* h1   = (_Float16*)(ws + 28 * MB + 65536);   // 2 MB

    const int NW = B_DIM * H_DIM;       // 1048576
    const int NWIH = G4H * H_DIM;       // 1048576
    const int NFC = V_DIM * H_DIM;      // 32768

    cvt_f32_f16<<<(NW + 255) / 256, 256, 0, stream>>>(z, zf, NW);
    cvt_f32_f16<<<(NWIH + 255) / 256, 256, 0, stream>>>(w_ih, wihf, NWIH);
    cvt_f32_f16<<<(NWIH + 255) / 256, 256, 0, stream>>>(w_hh, whhf, NWIH);
    cvt_f32_f16<<<(NFC + 255) / 256, 256, 0, stream>>>(fc_w, fcwf, NFC);
    init_state<<<(NW + 255) / 256, 256, 0, stream>>>(c, h0, NW);

    xgates_gemm<<<2048, 256, 0, stream>>>(zf, wihf, b_ih, b_hh, xg);

    for (int t = 0; t < T_DIM; ++t) {
        const _Float16* h_in = (t & 1) ? h1 : h0;
        _Float16* h_out      = (t & 1) ? h0 : h1;
        lstm_step<<<512, 256, 0, stream>>>(h_in, whhf, xg, c, h_out);
        fc_step<<<64, 256, 0, stream>>>(h_out, fcwf, fc_b, out, t);
    }
}
